// SimpleCodeBook_17300128268648
// MI455X (gfx1250) — matrix operand order repaired and checked by reading, not compiled, not run on hardware
//
#include <hip/hip_runtime.h>
#include <hip/hip_bf16.h>

// ---- problem constants ----
#define HH 8
#define BB 8192
#define CC 1024
#define DD 512

typedef __attribute__((ext_vector_type(16))) __bf16 v16bf;
typedef __attribute__((ext_vector_type(8)))  __bf16 v8bf;
typedef __attribute__((ext_vector_type(4)))  __bf16 v4bf;
typedef __attribute__((ext_vector_type(8)))  float  v8f;

// Split 8 fp32 values into bf16 hi (RNE) and bf16 lo (residual) chunks.
__device__ __forceinline__ void cvt8(float4 a, float4 b, v8bf& h, v8bf& l) {
    float f[8] = {a.x, a.y, a.z, a.w, b.x, b.y, b.z, b.w};
#pragma unroll
    for (int i = 0; i < 8; ++i) {
        __bf16 hb = (__bf16)f[i];
        h[i] = hb;
        l[i] = (__bf16)(f[i] - (float)hb);
    }
}

// ---------------- Kernel 0: row squared norms for x (wave per row of 512 f32) ----------------
__global__ void rownorm_kernel(const float* __restrict__ src, float* __restrict__ dst, int nrows) {
    const int wave = blockIdx.x * (blockDim.x >> 5) + (threadIdx.x >> 5);
    const int lane = threadIdx.x & 31;
    if (wave >= nrows) return;
    const float4* r = (const float4*)(src + (size_t)wave * DD);
    float s = 0.0f;
#pragma unroll
    for (int i = 0; i < 4; ++i) {
        float4 v = r[i * 32 + lane];
        s += v.x * v.x + v.y * v.y + v.z * v.z + v.w * v.w;
    }
#pragma unroll
    for (int off = 16; off; off >>= 1) s += __shfl_xor(s, off, 32);
    if (lane == 0) dst[wave] = s;
}

// ---------------- Kernel 1: embed prep: e2 + bf16 hi/lo planes (wave per row) ----------------
__global__ void embed_prep_kernel(const float* __restrict__ embed, float* __restrict__ e2,
                                  __bf16* __restrict__ ehi, __bf16* __restrict__ elo) {
    const int row  = blockIdx.x * (blockDim.x >> 5) + (threadIdx.x >> 5);  // 0..8191
    const int lane = threadIdx.x & 31;
    const float4* r = (const float4*)(embed + (size_t)row * DD);
    float s = 0.0f;
#pragma unroll
    for (int i = 0; i < 4; ++i) {
        float4 v = r[i * 32 + lane];
        s += v.x * v.x + v.y * v.y + v.z * v.z + v.w * v.w;
        float f[4] = {v.x, v.y, v.z, v.w};
        v4bf h4, l4;
#pragma unroll
        for (int j = 0; j < 4; ++j) {
            __bf16 hb = (__bf16)f[j];
            h4[j] = hb;
            l4[j] = (__bf16)(f[j] - (float)hb);
        }
        const size_t off = (size_t)row * DD + ((size_t)i * 32 + lane) * 4;
        *(v4bf*)(ehi + off) = h4;
        *(v4bf*)(elo + off) = l4;
    }
#pragma unroll
    for (int off = 16; off; off >>= 1) s += __shfl_xor(s, off, 32);
    if (lane == 0) e2[row] = s;
}

// ---------------- Kernel 2: batched GEMM + distance epilogue ----------------
// grid = (C/128, B/128, H); block = 128 threads (4 waves).
// Each wave owns 32 rows (two 16-row A fragments); block tile = 128x128.
// Double-buffered LDS slab + software-pipelined A loads hide global latency
// behind the 48 WMMAs per K-step.
__global__ void __launch_bounds__(128)
gemm_dist_kernel(const float* __restrict__ x,
                 const __bf16* __restrict__ ehi, const __bf16* __restrict__ elo,
                 const float* __restrict__ x2g, const float* __restrict__ e2g,
                 float* __restrict__ distOut) {
    const int h    = blockIdx.z;
    const int rb   = blockIdx.y;   // 0..63 (128-row blocks)
    const int cb   = blockIdx.x;   // 0..7  (128-col blocks)
    const int tid  = threadIdx.x;
    const int wave = tid >> 5;
    const int lane = tid & 31;

    __shared__ __bf16 sBhi[2][128 * 32];   // embed slab, bf16 hi plane: [buf][col][k]
    __shared__ __bf16 sBlo[2][128 * 32];   // embed slab, bf16 lo plane

    v8f acc0[8], acc1[8];
#pragma unroll
    for (int i = 0; i < 8; ++i) {
        acc0[i] = (v8f){0, 0, 0, 0, 0, 0, 0, 0};
        acc1[i] = (v8f){0, 0, 0, 0, 0, 0, 0, 0};
    }

    const int rowBase = rb * 128 + wave * 32;   // first of this wave's 32 rows within h
    const float* xrow0 = x + ((size_t)h * BB + rowBase + (lane & 15)) * DD;
    const float* xrow1 = xrow0 + (size_t)16 * DD;
    const int khalf = (lane >> 4) * 16;

    // LDS fill mapping: one thread per column, 32 k (64B per plane) each
    const size_t ebase = ((size_t)h * CC + cb * 128 + tid) * DD;

    auto fill = [&](int buf, int ks) {
        const uint4* gh = (const uint4*)(ehi + ebase + ks * 32);
        const uint4* gl = (const uint4*)(elo + ebase + ks * 32);
        uint4* sh = (uint4*)&sBhi[buf][tid * 32];
        uint4* sl = (uint4*)&sBlo[buf][tid * 32];
#pragma unroll
        for (int i = 0; i < 4; ++i) { sh[i] = gh[i]; sl[i] = gl[i]; }
    };
    auto loadAraw = [&](int ks, float4* r) {
        const float* p0 = xrow0 + ks * 32 + khalf;
        const float* p1 = xrow1 + ks * 32 + khalf;
        r[0] = *(const float4*)(p0);      r[1] = *(const float4*)(p0 + 4);
        r[2] = *(const float4*)(p0 + 8);  r[3] = *(const float4*)(p0 + 12);
        r[4] = *(const float4*)(p1);      r[5] = *(const float4*)(p1 + 4);
        r[6] = *(const float4*)(p1 + 8);  r[7] = *(const float4*)(p1 + 12);
    };

    float4 ra[8];
    fill(0, 0);          // prologue: stage slab 0
    loadAraw(0, ra);     // prologue: stage A raw for step 0

    for (int ks = 0; ks < DD / 32; ++ks) {
        const int p = ks & 1;
        __syncthreads();                       // buf[p] ready; buf[p^1] readers done
        if (ks + 1 < DD / 32) fill(p ^ 1, ks + 1);   // overlap next slab with compute

        // convert this step's A, then prefetch next step's raw A
        v16bf Ah0, Al0, Ah1, Al1;
        cvt8(ra[0], ra[1], ((v8bf*)&Ah0)[0], ((v8bf*)&Al0)[0]);
        cvt8(ra[2], ra[3], ((v8bf*)&Ah0)[1], ((v8bf*)&Al0)[1]);
        cvt8(ra[4], ra[5], ((v8bf*)&Ah1)[0], ((v8bf*)&Al1)[0]);
        cvt8(ra[6], ra[7], ((v8bf*)&Ah1)[1], ((v8bf*)&Al1)[1]);
        if (ks + 1 < DD / 32) loadAraw(ks + 1, ra);

        const int bK = (lane >> 4) * 16;   // B 32x16: lanes0-15 K=0..15, lanes16-31 K=16..31
#pragma unroll
        for (int tt = 0; tt < 8; ++tt) {
            const int colLocal = tt * 16 + (lane & 15);
            v16bf Bh = *(const v16bf*)&sBhi[p][colLocal * 32 + bK];
            v16bf Bl = *(const v16bf*)&sBlo[p][colLocal * 32 + bK];
            // 3xBF16 split, interleaved across the two row-sets for XDL spacing
            acc0[tt] = __builtin_amdgcn_wmma_f32_16x16x32_bf16(false, Ah0, false, Bh, (short)0, acc0[tt], false, false);
            acc1[tt] = __builtin_amdgcn_wmma_f32_16x16x32_bf16(false, Ah1, false, Bh, (short)0, acc1[tt], false, false);
            acc0[tt] = __builtin_amdgcn_wmma_f32_16x16x32_bf16(false, Ah0, false, Bl, (short)0, acc0[tt], false, false);
            acc1[tt] = __builtin_amdgcn_wmma_f32_16x16x32_bf16(false, Ah1, false, Bl, (short)0, acc1[tt], false, false);
            acc0[tt] = __builtin_amdgcn_wmma_f32_16x16x32_bf16(false, Al0, false, Bh, (short)0, acc0[tt], false, false);
            acc1[tt] = __builtin_amdgcn_wmma_f32_16x16x32_bf16(false, Al1, false, Bh, (short)0, acc1[tt], false, false);
        }
    }

    // Epilogue: dist = -sqrt(max(x2 + e2 - 2*xy, 0)); C/D layout:
    // VGPR j, lanes0-15 -> M=j, N=lane; lanes16-31 -> M=8+j, N=lane-16.
    const int rowHalf = (lane >> 4) * 8;
#pragma unroll
    for (int tt = 0; tt < 8; ++tt) {
        const int col = cb * 128 + tt * 16 + (lane & 15);
        const float e2v = e2g[h * CC + col];
#pragma unroll
        for (int j = 0; j < 8; ++j) {
            const int row0 = rowBase + rowHalf + j;
            float d2a = x2g[h * BB + row0] + e2v - 2.0f * acc0[tt][j];
            d2a = d2a > 0.0f ? d2a : 0.0f;
            distOut[(((size_t)h * BB + row0) * CC) + col] = -sqrtf(d2a);
            const int row1 = row0 + 16;
            float d2b = x2g[h * BB + row1] + e2v - 2.0f * acc1[tt][j];
            d2b = d2b > 0.0f ? d2b : 0.0f;
            distOut[(((size_t)h * BB + row1) * CC) + col] = -sqrtf(d2b);
        }
    }
}

// ---------------- Kernel 3: masked argmax per row + gather quantize ----------------
__global__ void argmax_gather_kernel(const float* __restrict__ dist,
                                     const float* __restrict__ embed,
                                     const unsigned char* __restrict__ valid,
                                     float* __restrict__ quant, int* __restrict__ ind) {
    const int row  = blockIdx.x * (blockDim.x >> 5) + (threadIdx.x >> 5);  // 0..65535
    const int lane = threadIdx.x & 31;
    const int h = row >> 13;

    const float* drow = dist + (size_t)row * CC;
    const unsigned char* vrow = valid + h * CC;

    float best = -3.0e38f;
    int bi = 0;
    for (int c = lane; c < CC; c += 32) {
        float v = drow[c];
        v = vrow[c] ? v : -1.0e10f;          // MASK_FILL
        if (v > best || (v == best && c < bi)) { best = v; bi = c; }
    }
#pragma unroll
    for (int off = 16; off; off >>= 1) {
        float ov = __shfl_xor(best, off, 32);
        int   oi = __shfl_xor(bi, off, 32);
        if (ov > best || (ov == best && oi < bi)) { best = ov; bi = oi; }
    }
    if (lane == 0) ind[row] = bi;

    // gather: quantize[row,:] = embed[h, bi, :]
    const float4* erow = (const float4*)(embed + ((size_t)h * CC + bi) * DD);
    float4* qrow = (float4*)(quant + (size_t)row * DD);
#pragma unroll
    for (int i = lane; i < DD / 4; i += 32) qrow[i] = erow[i];
}

extern "C" void kernel_launch(void* const* d_in, const int* in_sizes, int n_in,
                              void* d_out, int out_size, void* d_ws, size_t ws_size,
                              hipStream_t stream) {
    const float*         x     = (const float*)d_in[0];
    const float*         embed = (const float*)d_in[1];
    const unsigned char* valid = (const unsigned char*)d_in[2];

    float* out = (float*)d_out;
    constexpr size_t QN = (size_t)HH * BB * DD;   // 33,554,432
    constexpr size_t IN = (size_t)HH * BB;        // 65,536
    float* quant = out;
    int*   ind   = (int*)(out + QN);
    float* dist  = out + QN + IN;

    // workspace: x2 (256KB) | e2 (32KB) | ehi (8MB bf16) | elo (8MB bf16)
    float*  x2  = (float*)d_ws;
    float*  e2  = x2 + (size_t)HH * BB;
    __bf16* ehi = (__bf16*)(e2 + (size_t)HH * CC);
    __bf16* elo = ehi + (size_t)HH * CC * DD;

    // 1) x row norms; embed prep (e2 + bf16 hi/lo planes)
    rownorm_kernel<<<(HH * BB) / 8, 256, 0, stream>>>(x, x2, HH * BB);
    embed_prep_kernel<<<(HH * CC) / 8, 256, 0, stream>>>(embed, e2, ehi, elo);

    // 2) batched GEMM + distance epilogue (4 waves/block, 128x128 tile, double-buffered)
    dim3 grid(CC / 128, BB / 128, HH);
    gemm_dist_kernel<<<grid, 128, 0, stream>>>(x, ehi, elo, x2, e2, dist);

    // 3) masked argmax + gather
    argmax_gather_kernel<<<(HH * BB) / 8, 256, 0, stream>>>(dist, embed, valid, quant, ind);
}
